// LCF_ATEPC_27470610825705
// MI455X (gfx1250) — matrix operand order repaired and checked by reading, not compiled, not run on hardware
//
#include <hip/hip_runtime.h>
#include <hip/hip_bf16.h>
#include <math.h>

// ---------------------------------------------------------------------------
// LCF-ATEPC (dual BERT + LCF head) for MI455X / gfx1250, wave32 + WMMA.
// All heavy GEMMs run through one v_wmma_f32_16x16x32_f16 tiled kernel with a
// software-pipelined (register double-buffered) global->LDS staging path.
// ---------------------------------------------------------------------------

typedef _Float16 v16h  __attribute__((ext_vector_type(16)));
typedef float    v8f   __attribute__((ext_vector_type(8)));
typedef _Float16 half8 __attribute__((ext_vector_type(8)));
typedef _Float16 half4 __attribute__((ext_vector_type(4)));

static constexpr int  BB  = 16;
static constexpr int  SS  = 256;
static constexpr int  HD  = 768;
static constexpr int  NHH = 12;
static constexpr int  FF  = 3072;
static constexpr int  TOK = BB * SS;                    // 4096
static constexpr long long HHs  = (long long)TOK * HD;  // 3,145,728
static constexpr long long BIGs = (long long)TOK * FF;  // scores OR ffn-hidden

// ---------------------------------------------------------------------------
// Generic WMMA GEMM:  C = epi(alpha * A*B + bias + cbias + resid)
//   A: fp32 [M,K] row-major (row stride lda), per-z offset zo*sAo + zi*sAi
//   B: fp32; BT=false -> B[k*ldb+n], BT=true -> B[n*ldb+k]
//   EPI: 0 = none, 1 = exact GELU, 2 = tanh
// Tiles: BM=128 BN=128 BK=64, 256 threads = 8 waves (4 in M x 2 in N),
// wave tile 32x64 -> 8 accumulators of v_wmma_f32_16x16x32_f16.
// Pipeline: fetch K-tile t+1 into float4 regs while WMMA consumes tile t from
// LDS; fp32->f16 conversion deferred to the LDS-store phase.
// ---------------------------------------------------------------------------
template<int EPI, bool BT>
__global__ __launch_bounds__(256) void wmma_gemm_k(
    const float* __restrict__ A, long long sAo, long long sAi, int lda,
    const float* __restrict__ Bm, long long sBo, long long sBi, int ldb,
    const float* __restrict__ bias,
    const float* __restrict__ cbias, long long sCb,
    const float* __restrict__ resid,
    float* __restrict__ C, long long sCo, long long sCi, int ldc,
    int M, int N, int K, int NI, float alpha)
{
  __shared__ _Float16 As[128][72];   // +8 half pad: conflict-free frag loads
  __shared__ _Float16 Bs[128][72];   // stored K-major: Bs[n][k]

  const int tid = threadIdx.x;
  const int z  = blockIdx.z;
  const int zo = z / NI, zi = z - zo * NI;
  const long long aofs = zo * sAo + zi * sAi;
  const long long bofs = zo * sBo + zi * sBi;
  const long long cofs = zo * sCo + zi * sCi;
  const int blockRow = blockIdx.y * 128;
  const int blockCol = blockIdx.x * 128;
  const int wave = tid >> 5;
  const int lane = tid & 31;
  const int g  = lane >> 4;       // half-wave group (WMMA layout)
  const int mr = lane & 15;
  const int wm = wave & 3;        // wave position in M (4)
  const int wn = wave >> 2;       // wave position in N (2)

  v8f acc[2][4];
#pragma unroll
  for (int i = 0; i < 2; ++i)
#pragma unroll
    for (int j = 0; j < 4; ++j)
#pragma unroll
      for (int e = 0; e < 8; ++e) acc[i][j][e] = 0.0f;

  const bool bvec = BT ? true : (((N & 3) == 0) && ((ldb & 3) == 0));

  // ---- register staging buffers (raw fp32, converted at LDS-store time) ---
  float4 ga[8];      // A tile slices
  float4 gb[8];      // B tile slices (BT / vector path)
  float  gbs[32];    // B tile (narrow-N scalar path)

  // fetch K-tile starting at k0 into registers (global loads only)
  auto fetch = [&](int k0) {
#pragma unroll
    for (int it = 0; it < 8; ++it) {
      int lin = tid + it * 256;
      int row = lin >> 4, c4 = lin & 15;
      int gr = blockRow + row;
      float4 f = {0.0f, 0.0f, 0.0f, 0.0f};
      if (gr < M)
        f = *reinterpret_cast<const float4*>(
            A + aofs + (long long)gr * lda + k0 + c4 * 4);
      ga[it] = f;
    }
    if (BT) {
#pragma unroll
      for (int it = 0; it < 8; ++it) {
        int lin = tid + it * 256;
        int n = lin >> 4, k4 = lin & 15;
        int gn = blockCol + n;
        float4 f = {0.0f, 0.0f, 0.0f, 0.0f};
        if (gn < N)
          f = *reinterpret_cast<const float4*>(
              Bm + bofs + (long long)gn * ldb + k0 + k4 * 4);
        gb[it] = f;
      }
    } else if (bvec) {
#pragma unroll
      for (int it = 0; it < 8; ++it) {
        int lin = tid + it * 256;
        int kk = lin >> 5, n4 = lin & 31;
        int col0 = blockCol + n4 * 4;
        float4 f = {0.0f, 0.0f, 0.0f, 0.0f};
        if (col0 < N)
          f = *reinterpret_cast<const float4*>(
              Bm + bofs + (long long)(k0 + kk) * ldb + col0);
        gb[it] = f;
      }
    } else {
#pragma unroll
      for (int it = 0; it < 32; ++it) {
        int lin = tid + it * 256;
        int kk = lin >> 7, n = lin & 127;
        int gn = blockCol + n;
        gbs[it] = (gn < N) ? Bm[bofs + (long long)(k0 + kk) * ldb + gn] : 0.0f;
      }
    }
  };

  // convert staged registers to f16 and store into LDS
  auto stage = [&]() {
#pragma unroll
    for (int it = 0; it < 8; ++it) {
      int lin = tid + it * 256;
      int row = lin >> 4, c4 = lin & 15;
      half4 hv;
      hv[0] = (_Float16)ga[it].x; hv[1] = (_Float16)ga[it].y;
      hv[2] = (_Float16)ga[it].z; hv[3] = (_Float16)ga[it].w;
      *reinterpret_cast<half4*>(&As[row][c4 * 4]) = hv;
    }
    if (BT) {
#pragma unroll
      for (int it = 0; it < 8; ++it) {
        int lin = tid + it * 256;
        int n = lin >> 4, k4 = lin & 15;
        half4 hv;
        hv[0] = (_Float16)gb[it].x; hv[1] = (_Float16)gb[it].y;
        hv[2] = (_Float16)gb[it].z; hv[3] = (_Float16)gb[it].w;
        *reinterpret_cast<half4*>(&Bs[n][k4 * 4]) = hv;
      }
    } else if (bvec) {
#pragma unroll
      for (int it = 0; it < 8; ++it) {
        int lin = tid + it * 256;
        int kk = lin >> 5, n4 = lin & 31;
        Bs[n4 * 4 + 0][kk] = (_Float16)gb[it].x;
        Bs[n4 * 4 + 1][kk] = (_Float16)gb[it].y;
        Bs[n4 * 4 + 2][kk] = (_Float16)gb[it].z;
        Bs[n4 * 4 + 3][kk] = (_Float16)gb[it].w;
      }
    } else {
#pragma unroll
      for (int it = 0; it < 32; ++it) {
        int lin = tid + it * 256;
        int kk = lin >> 7, n = lin & 127;
        Bs[n][kk] = (_Float16)gbs[it];
      }
    }
  };

  fetch(0);
  for (int k0 = 0; k0 < K; k0 += 64) {
    stage();
    __syncthreads();
    if (k0 + 64 < K) fetch(k0 + 64);   // next tile in flight during WMMA

    // ---- WMMA over the staged tile ---------------------------------------
#pragma unroll
    for (int ks = 0; ks < 64; ks += 32) {
      v16h af[2], bf[4];
#pragma unroll
      for (int ti = 0; ti < 2; ++ti) {
        int row = wm * 32 + ti * 16 + mr;
        // A frag: elem j -> K = ks + g*8 + j (j<8), ks + 16 + g*8 + (j-8)
        half8 lo = *reinterpret_cast<const half8*>(&As[row][ks + g * 8]);
        half8 hi = *reinterpret_cast<const half8*>(&As[row][ks + 16 + g * 8]);
#pragma unroll
        for (int j = 0; j < 8; ++j) { af[ti][j] = lo[j]; af[ti][8 + j] = hi[j]; }
      }
#pragma unroll
      for (int tj = 0; tj < 4; ++tj) {
        int col = wn * 64 + tj * 16 + mr;
        // B frag: elem j -> K = ks + g*8 + j (j<8), ks + 16 + g*8 + (j-8)
        half8 lo = *reinterpret_cast<const half8*>(&Bs[col][ks + g * 8]);
        half8 hi = *reinterpret_cast<const half8*>(&Bs[col][ks + 16 + g * 8]);
#pragma unroll
        for (int j = 0; j < 8; ++j) { bf[tj][j] = lo[j]; bf[tj][8 + j] = hi[j]; }
      }
#pragma unroll
      for (int ti = 0; ti < 2; ++ti)
#pragma unroll
        for (int tj = 0; tj < 4; ++tj)
          acc[ti][tj] = __builtin_amdgcn_wmma_f32_16x16x32_f16(
              false, af[ti], false, bf[tj], (short)0, acc[ti][tj], false, false);
    }
    __syncthreads();
  }

  // ---- epilogue: C layout VGPR i -> row = base + g*8 + i, col = base + mr --
#pragma unroll
  for (int ti = 0; ti < 2; ++ti) {
#pragma unroll
    for (int tj = 0; tj < 4; ++tj) {
      int nn = blockCol + wn * 64 + tj * 16 + mr;
      if (nn >= N) continue;
#pragma unroll
      for (int i = 0; i < 8; ++i) {
        int mm = blockRow + wm * 32 + ti * 16 + g * 8 + i;
        if (mm >= M) continue;
        float v = acc[ti][tj][i] * alpha;
        if (bias)  v += bias[nn];
        if (cbias) v += cbias[zo * sCb + nn];
        long long oidx = cofs + (long long)mm * ldc + nn;
        if (resid) v += resid[oidx];
        if (EPI == 1) v = 0.5f * v * (1.0f + erff(v * 0.70710678118654752440f));
        else if (EPI == 2) v = tanhf(v);
        C[oidx] = v;
      }
    }
  }
}

// ---------------------------------------------------------------------------
// Row softmax, 256 cols, one wave32 per row (8 rows / block).
// ---------------------------------------------------------------------------
__global__ __launch_bounds__(256) void softmax256_k(float* __restrict__ s)
{
  int row = blockIdx.x * 8 + (threadIdx.x >> 5);
  int l = threadIdx.x & 31;
  float4* p = reinterpret_cast<float4*>(s + (long long)row * 256);
  float4 a = p[l], b = p[l + 32];
  float mx = fmaxf(fmaxf(fmaxf(a.x, a.y), fmaxf(a.z, a.w)),
                   fmaxf(fmaxf(b.x, b.y), fmaxf(b.z, b.w)));
#pragma unroll
  for (int off = 16; off > 0; off >>= 1) mx = fmaxf(mx, __shfl_xor(mx, off, 32));
  a.x = expf(a.x - mx); a.y = expf(a.y - mx); a.z = expf(a.z - mx); a.w = expf(a.w - mx);
  b.x = expf(b.x - mx); b.y = expf(b.y - mx); b.z = expf(b.z - mx); b.w = expf(b.w - mx);
  float sm = a.x + a.y + a.z + a.w + b.x + b.y + b.z + b.w;
#pragma unroll
  for (int off = 16; off > 0; off >>= 1) sm += __shfl_xor(sm, off, 32);
  float inv = 1.0f / sm;
  a.x *= inv; a.y *= inv; a.z *= inv; a.w *= inv;
  b.x *= inv; b.y *= inv; b.z *= inv; b.w *= inv;
  p[l] = a; p[l + 32] = b;
}

// ---------------------------------------------------------------------------
// LayerNorm over H=768, one block (256 thr) per row; eps = 1e-12.
// ---------------------------------------------------------------------------
__global__ __launch_bounds__(256) void ln768_k(
    const float* __restrict__ in, const float* __restrict__ gg,
    const float* __restrict__ bb, float* __restrict__ out)
{
  int row = blockIdx.x, t = threadIdx.x;
  long long o = (long long)row * 768;
  float v0 = in[o + t], v1 = in[o + t + 256], v2 = in[o + t + 512];
  __shared__ float r1[256], r2[256];
  r1[t] = v0 + v1 + v2;
  r2[t] = v0 * v0 + v1 * v1 + v2 * v2;
  __syncthreads();
  for (int off = 128; off > 0; off >>= 1) {
    if (t < off) { r1[t] += r1[t + off]; r2[t] += r2[t + off]; }
    __syncthreads();
  }
  float mean = r1[0] * (1.0f / 768.0f);
  float var  = r2[0] * (1.0f / 768.0f) - mean * mean;
  float rstd = rsqrtf(var + 1e-12f);
  out[o + t]       = (v0 - mean) * rstd * gg[t]       + bb[t];
  out[o + t + 256] = (v1 - mean) * rstd * gg[t + 256] + bb[t + 256];
  out[o + t + 512] = (v2 - mean) * rstd * gg[t + 512] + bb[t + 512];
}

// Embedding gather + LayerNorm fused, one block per token.
__global__ __launch_bounds__(256) void embed_ln_k(
    const int* __restrict__ ids, const float* __restrict__ ew,
    const float* __restrict__ ep, const float* __restrict__ gg,
    const float* __restrict__ bb, float* __restrict__ out)
{
  int tok = blockIdx.x, t = threadIdx.x;
  int s = tok & (SS - 1);
  long long wb = (long long)ids[tok] * 768;
  int pb = s * 768;
  float v0 = ew[wb + t]       + ep[pb + t];
  float v1 = ew[wb + t + 256] + ep[pb + t + 256];
  float v2 = ew[wb + t + 512] + ep[pb + t + 512];
  __shared__ float r1[256], r2[256];
  r1[t] = v0 + v1 + v2;
  r2[t] = v0 * v0 + v1 * v1 + v2 * v2;
  __syncthreads();
  for (int off = 128; off > 0; off >>= 1) {
    if (t < off) { r1[t] += r1[t + off]; r2[t] += r2[t + off]; }
    __syncthreads();
  }
  float mean = r1[0] * (1.0f / 768.0f);
  float var  = r2[0] * (1.0f / 768.0f) - mean * mean;
  float rstd = rsqrtf(var + 1e-12f);
  long long o = (long long)tok * 768;
  out[o + t]       = (v0 - mean) * rstd * gg[t]       + bb[t];
  out[o + t + 256] = (v1 - mean) * rstd * gg[t + 256] + bb[t + 256];
  out[o + t + 512] = (v2 - mean) * rstd * gg[t + 512] + bb[t + 512];
}

// mask bias: (1 - attn_mask) * -10000
__global__ void maskbias_k(const int* __restrict__ m, float* __restrict__ mb, int n)
{
  int i = blockIdx.x * blockDim.x + threadIdx.x;
  if (i < n) mb[i] = (1.0f - (float)m[i]) * -10000.0f;
}

// first SEP position per row -> zero ids after it (mask_after_sep)
__global__ __launch_bounds__(256) void sep_localids_k(
    const int* __restrict__ ids, int* __restrict__ lids)
{
  int b = blockIdx.x, s = threadIdx.x;
  int id = ids[b * SS + s];
  __shared__ int r[256];
  r[s] = (id == 102) ? s : 0x7FFFFFFF;
  __syncthreads();
  for (int off = 128; off > 0; off >>= 1) {
    if (s < off && r[s + off] < r[s]) r[s] = r[s + off];
    __syncthreads();
  }
  int sep = r[0];
  if (sep == 0x7FFFFFFF) sep = 0;   // argmax of all-false == 0
  lids[b * SS + s] = (s <= sep) ? id : 0;
}

// packed positions (cumsum(valid)-1 where valid, else -1)
__global__ void pack_pos_k(const int* __restrict__ valid, int* __restrict__ pos)
{
  int b = blockIdx.x;
  if (threadIdx.x == 0) {
    int c = 0;
    for (int s = 0; s < SS; ++s) {
      int vf = valid[b * SS + s];
      pos[b * SS + s] = (vf == 1) ? c : -1;
      if (vf == 1) c++;
    }
  }
}

// scatter rows into packed buffer (out pre-zeroed)
__global__ __launch_bounds__(256) void pack_scatter_k(
    const float* __restrict__ x, const int* __restrict__ pos, float* __restrict__ out)
{
  int tok = blockIdx.x;
  int p = pos[tok];
  if (p < 0) return;
  int b = tok >> 8;
  const float* src = x + (long long)tok * 768;
  float* dst = out + (long long)(b * SS + p) * 768;
  dst[threadIdx.x]       = src[threadIdx.x];
  dst[threadIdx.x + 256] = src[threadIdx.x + 256];
  dst[threadIdx.x + 512] = src[threadIdx.x + 512];
}

// cat = [g, loc*cdw, loc*cdm]
__global__ __launch_bounds__(256) void build_cat_k(
    const float* __restrict__ g, const float* __restrict__ loc,
    const float* __restrict__ cdm, const float* __restrict__ cdw,
    float* __restrict__ cat)
{
  int tok = blockIdx.x;
  float wm = cdm[tok], ww = cdw[tok];
  long long ib = (long long)tok * 768, ob = (long long)tok * 2304;
#pragma unroll
  for (int j = 0; j < 3; ++j) {
    int h = threadIdx.x + j * 256;
    float lv = loc[ib + h];
    cat[ob + h]        = g[ib + h];
    cat[ob + 768 + h]  = lv * ww;
    cat[ob + 1536 + h] = lv * wm;
  }
}

// ---------------------------------------------------------------------------
// Host-side dispatch
// ---------------------------------------------------------------------------
static void gemm(hipStream_t st, int epi, bool bt,
                 const float* A, long long sAo, long long sAi, int lda,
                 const float* Bm, long long sBo, long long sBi, int ldb,
                 const float* bias, const float* cbias, long long sCb,
                 const float* resid,
                 float* C, long long sCo, long long sCi, int ldc,
                 int M, int N, int K, int Z, int NI, float alpha)
{
  dim3 grid((N + 127) / 128, (M + 127) / 128, Z), block(256);
  if (bt)
    wmma_gemm_k<0, true><<<grid, block, 0, st>>>(A, sAo, sAi, lda, Bm, sBo, sBi, ldb,
        bias, cbias, sCb, resid, C, sCo, sCi, ldc, M, N, K, NI, alpha);
  else if (epi == 1)
    wmma_gemm_k<1, false><<<grid, block, 0, st>>>(A, sAo, sAi, lda, Bm, sBo, sBi, ldb,
        bias, cbias, sCb, resid, C, sCo, sCi, ldc, M, N, K, NI, alpha);
  else if (epi == 2)
    wmma_gemm_k<2, false><<<grid, block, 0, st>>>(A, sAo, sAi, lda, Bm, sBo, sBi, ldb,
        bias, cbias, sCb, resid, C, sCo, sCi, ldc, M, N, K, NI, alpha);
  else
    wmma_gemm_k<0, false><<<grid, block, 0, st>>>(A, sAo, sAi, lda, Bm, sBo, sBi, ldb,
        bias, cbias, sCb, resid, C, sCo, sCi, ldc, M, N, K, NI, alpha);
}

// full multi-head attention + residual/LN/FFN per layer (weights at d_in[base..])
static void bert_forward(hipStream_t st, void* const* d_in, const int* ids,
                         const float* cbias, float* x, float* q, float* k,
                         float* v, float* tmp, float* ctx, float* big)
{
  auto F = [&](int i) { return (const float*)d_in[i]; };
  const long long SH = (long long)SS * HD;       // per-batch activation stride
  const long long SSQ = (long long)SS * SS;      // per-head score stride
  embed_ln_k<<<TOK, 256, 0, st>>>(ids, F(5), F(6), F(7), F(8), x);
  for (int l = 0; l < 4; ++l) {
    int b0 = 9 + 16 * l;
    // Q,K,V projections
    gemm(st, 0, false, x, 0, 0, HD, F(b0 + 0), 0, 0, HD, F(b0 + 1), nullptr, 0, nullptr,
         q, 0, 0, HD, TOK, HD, HD, 1, 1, 1.0f);
    gemm(st, 0, false, x, 0, 0, HD, F(b0 + 2), 0, 0, HD, F(b0 + 3), nullptr, 0, nullptr,
         k, 0, 0, HD, TOK, HD, HD, 1, 1, 1.0f);
    gemm(st, 0, false, x, 0, 0, HD, F(b0 + 4), 0, 0, HD, F(b0 + 5), nullptr, 0, nullptr,
         v, 0, 0, HD, TOK, HD, HD, 1, 1, 1.0f);
    // scores = q @ k^T / 8 + maskbias  (z = b*NH + h)
    gemm(st, 0, true, q, SH, 64, HD, k, SH, 64, HD, nullptr, cbias, SS, nullptr,
         big, (long long)NHH * SSQ, SSQ, SS, SS, SS, 64, BB * NHH, NHH, 0.125f);
    softmax256_k<<<(BB * NHH * SS) / 8, 256, 0, st>>>(big);
    // ctx = P @ v
    gemm(st, 0, false, big, (long long)NHH * SSQ, SSQ, SS, v, SH, 64, HD,
         nullptr, nullptr, 0, nullptr, ctx, SH, 64, HD, SS, 64, SS, BB * NHH, NHH, 1.0f);
    // O projection + residual, then LN1
    gemm(st, 0, false, ctx, 0, 0, HD, F(b0 + 6), 0, 0, HD, F(b0 + 7), nullptr, 0, x,
         tmp, 0, 0, HD, TOK, HD, HD, 1, 1, 1.0f);
    ln768_k<<<TOK, 256, 0, st>>>(tmp, F(b0 + 8), F(b0 + 9), x);
    // FFN: gelu(x@W1+b1) @ W2 + b2 + residual, then LN2
    gemm(st, 1, false, x, 0, 0, HD, F(b0 + 10), 0, 0, FF, F(b0 + 11), nullptr, 0, nullptr,
         big, 0, 0, FF, TOK, FF, HD, 1, 1, 1.0f);
    gemm(st, 0, false, big, 0, 0, FF, F(b0 + 12), 0, 0, HD, F(b0 + 13), nullptr, 0, x,
         tmp, 0, 0, HD, TOK, HD, FF, 1, 1, 1.0f);
    ln768_k<<<TOK, 256, 0, st>>>(tmp, F(b0 + 14), F(b0 + 15), x);
  }
}

extern "C" void kernel_launch(void* const* d_in, const int* in_sizes, int n_in,
                              void* d_out, int out_size, void* d_ws, size_t ws_size,
                              hipStream_t stream)
{
  (void)in_sizes; (void)n_in; (void)out_size; (void)ws_size;
  const int*   ids   = (const int*)d_in[0];
  const int*   amask = (const int*)d_in[1];
  const int*   valid = (const int*)d_in[2];
  const float* cdm   = (const float*)d_in[3];
  const float* cdw   = (const float*)d_in[4];
  auto F = [&](int i) { return (const float*)d_in[i]; };
  float* out = (float*)d_out;

  // ---- workspace layout (floats) ----
  float* ws   = (float*)d_ws;
  float* x    = ws;
  float* q    = ws + 1 * HHs;
  float* k    = ws + 2 * HHs;
  float* v    = ws + 3 * HHs;
  float* tmp  = ws + 4 * HHs;     // also reused as cat_t
  float* ctx  = ws + 5 * HHs;
  float* gpk  = ws + 6 * HHs;
  float* lpk  = ws + 7 * HHs;
  float* smallp  = ws + 8 * HHs;
  float* mbias   = smallp;                      // B*S
  float* pooled  = smallp + 4096;               // B*H
  int*   ipos    = (int*)(smallp + 4096 + 12288);
  int*   lids    = ipos + TOK;
  float* big     = ws + 8 * HHs + 32768;        // scores OR ffn hidden (TOK*FF)
  float* cat     = big + BIGs;                  // TOK * 2304
  float* catt    = tmp;

  const long long SH  = (long long)SS * HD;
  const long long SSQ = (long long)SS * SS;

  // ---- prep ----
  maskbias_k<<<(BB * SS + 255) / 256, 256, 0, stream>>>(amask, mbias, BB * SS);
  sep_localids_k<<<BB, 256, 0, stream>>>(ids, lids);
  pack_pos_k<<<BB, 32, 0, stream>>>(valid, ipos);

  // ---- global BERT -> gpk ----
  bert_forward(stream, d_in, ids, mbias, x, q, k, v, tmp, ctx, big);
  hipMemsetAsync(gpk, 0, (size_t)HHs * sizeof(float), stream);
  pack_scatter_k<<<TOK, 256, 0, stream>>>(x, ipos, gpk);

  // ---- local BERT (sep-masked ids, all-ones mask -> no bias) -> lpk ----
  bert_forward(stream, d_in, lids, nullptr, x, q, k, v, tmp, ctx, big);
  hipMemsetAsync(lpk, 0, (size_t)HHs * sizeof(float), stream);
  pack_scatter_k<<<TOK, 256, 0, stream>>>(x, ipos, lpk);

  // ---- ate_logits = gpk @ Wc + bc  -> out[0 : B*S*6] ----
  gemm(stream, 0, false, gpk, 0, 0, HD, F(91), 0, 0, 6, F(92), nullptr, 0, nullptr,
       out, 0, 0, 6, TOK, 6, HD, 1, 1, 1.0f);

  // ---- LCF head ----
  build_cat_k<<<TOK, 256, 0, stream>>>(gpk, lpk, cdm, cdw, cat);
  gemm(stream, 0, false, cat, 0, 0, 3 * HD, F(85), 0, 0, HD, F(86), nullptr, 0, nullptr,
       catt, 0, 0, HD, TOK, HD, 3 * HD, 1, 1, 1.0f);

  // SA2 attention on catt, tanh on context
  gemm(stream, 0, false, catt, 0, 0, HD, F(79), 0, 0, HD, F(80), nullptr, 0, nullptr,
       q, 0, 0, HD, TOK, HD, HD, 1, 1, 1.0f);
  gemm(stream, 0, false, catt, 0, 0, HD, F(81), 0, 0, HD, F(82), nullptr, 0, nullptr,
       k, 0, 0, HD, TOK, HD, HD, 1, 1, 1.0f);
  gemm(stream, 0, false, catt, 0, 0, HD, F(83), 0, 0, HD, F(84), nullptr, 0, nullptr,
       v, 0, 0, HD, TOK, HD, HD, 1, 1, 1.0f);
  gemm(stream, 0, true, q, SH, 64, HD, k, SH, 64, HD, nullptr, nullptr, 0, nullptr,
       big, (long long)NHH * SSQ, SSQ, SS, SS, SS, 64, BB * NHH, NHH, 0.125f);
  softmax256_k<<<(BB * NHH * SS) / 8, 256, 0, stream>>>(big);
  gemm(stream, 2, false, big, (long long)NHH * SSQ, SSQ, SS, v, SH, 64, HD,
       nullptr, nullptr, 0, nullptr, ctx, SH, 64, HD, SS, 64, SS, BB * NHH, NHH, 1.0f);

  // pooled = tanh(sa_out[:,0] @ Wp + bp): rows of ctx at stride S*H
  gemm(stream, 2, false, ctx, 0, 0, (int)SH, F(87), 0, 0, HD, F(88), nullptr, 0, nullptr,
       pooled, 0, 0, HD, BB, HD, HD, 1, 1, 1.0f);
  // apc_logits = pooled @ Wd + bd -> out[B*S*6 : +B*3]
  gemm(stream, 0, false, pooled, 0, 0, HD, F(89), 0, 0, 3, F(90), nullptr, 0, nullptr,
       out + (long long)TOK * 6, 0, 0, 3, BB, 3, HD, 1, 1, 1.0f);
}
